// AttFlow_601295421903
// MI455X (gfx1250) — matrix operand order repaired and checked by reading, not compiled, not run on hardware
//
#include <hip/hip_runtime.h>
#include <hip/hip_bf16.h>

// CDNA5 / gfx1250 WMMA types
typedef __attribute__((ext_vector_type(16))) __bf16 v16bf;
typedef __attribute__((ext_vector_type(8)))  __bf16 v8bf;
typedef __attribute__((ext_vector_type(8)))  float  v8f;

constexpr int B_  = 32;
constexpr int LC  = 1024;
constexpr int LQ  = 128;
constexpr int DIM = 1024;

__device__ __forceinline__ __bf16 tobf(float f) { return (__bf16)f; }

// ---------------------------------------------------------------------------
// K1: cdot[b,c] = <context[b,c,:], w1>,  qdot[b,q] = <question[b,q,:], w2>
// 4 threads per row (aligned 4-lane groups) -> 576 blocks of memory-level
// parallelism; partials combined with shfl_xor within the group.
// ---------------------------------------------------------------------------
__global__ __launch_bounds__(256)
void k_rowdots(const float* __restrict__ ctx, const float* __restrict__ qry,
               const float* __restrict__ w,
               float* __restrict__ cdot, float* __restrict__ qdot)
{
    const int t    = blockIdx.x * 256 + threadIdx.x;
    const int row  = t >> 2;
    const int part = t & 3;

    const float4* x;
    const float4* wv;
    float* outp;
    int oidx;
    if (row < B_ * LC) {
        x    = (const float4*)(ctx + (size_t)row * DIM);
        wv   = (const float4*)(w);           // w1
        outp = cdot; oidx = row;
    } else {
        const int r2 = row - B_ * LC;        // < B_*LQ by grid sizing
        x    = (const float4*)(qry + (size_t)r2 * DIM);
        wv   = (const float4*)(w + DIM);     // w2
        outp = qdot; oidx = r2;
    }
    x  += part * 64;                          // quarter of the row (256 floats)
    wv += part * 64;

    float s = 0.f;
    #pragma unroll 4
    for (int i = 0; i < 64; ++i) {
        const float4 a = x[i], b = wv[i];
        s += a.x * b.x + a.y * b.y + a.z * b.z + a.w * b.w;
    }
    s += __shfl_xor(s, 1, 32);
    s += __shfl_xor(s, 2, 32);
    if (part == 0) outp[oidx] = s;
}

// ---------------------------------------------------------------------------
// K1b: one-time fp32 -> bf16 conversion of Q in both layouts:
//   Qbf [b][q][d]  (B-fragments for k_sim:  K = d contiguous)
//   QbfT[b][d][q]  (B-fragments for k_pv:   K = q contiguous)
// ---------------------------------------------------------------------------
__global__ __launch_bounds__(256)
void k_convq(const float* __restrict__ qry, __bf16* __restrict__ Qbf,
             __bf16* __restrict__ QbfT)
{
    const int b = blockIdx.y;
    const int d = blockIdx.x * 256 + threadIdx.x;
    const float* qb   = qry + (size_t)b * LQ * DIM + d;
    __bf16*      trow = QbfT + ((size_t)b * DIM + d) * LQ;

    #pragma unroll
    for (int ch = 0; ch < LQ / 16; ++ch) {
        v16bf tmp;
        #pragma unroll
        for (int qq = 0; qq < 16; ++qq) {
            const int q = ch * 16 + qq;
            const __bf16 bv = tobf(qb[(size_t)q * DIM]);
            Qbf[((size_t)b * LQ + q) * DIM + d] = bv;
            tmp[qq] = bv;
        }
        *(v16bf*)(trow + ch * 16) = tmp;   // 32B contiguous store
    }
}

// ---------------------------------------------------------------------------
// K2: sim tile (64 x 128) per block via v_wmma_f32_16x16x32_bf16,
//     + bias add + softmax over q, writes P (bf16 probs) and rowmax.
// Block: 128 threads = 4 waves; wave w owns rows [ctile*64 + w*16, +16).
// w3 is staged into LDS with CDNA5 async global->LDS copies (ASYNCcnt).
// ---------------------------------------------------------------------------
__global__ __launch_bounds__(128)
void k_sim(const float* __restrict__ ctx, const __bf16* __restrict__ Qbf,
           const float* __restrict__ w, const float* __restrict__ cdot,
           const float* __restrict__ qdot, float* __restrict__ rowmax,
           __bf16* __restrict__ P)
{
    const int ctile = blockIdx.x;    // 0..15
    const int b     = blockIdx.y;    // 0..31
    const int tid   = threadIdx.x;
    const int wave  = tid >> 5;
    const int lane  = tid & 31;
    const int half  = lane >> 4;     // 0|1
    const int lm    = lane & 15;

    __shared__ float w3s[DIM];
    __shared__ float qdots[LQ];
    __shared__ float cdots[64];

    // --- async global->LDS staging of w3 (4 KB): 128 lanes x 32 B ---
    {
        const unsigned ldsAddr = ((unsigned)(size_t)(&w3s[0])) + (unsigned)tid * 32u;
        const unsigned gOff    = (unsigned)tid * 32u;
        const float*   w3p     = w + 2 * DIM;
        asm volatile(
            "global_load_async_to_lds_b128 %0, %1, %2\n\t"
            "global_load_async_to_lds_b128 %0, %1, %2 offset:16\n\t"
            "s_wait_asynccnt 0"
            :: "v"(ldsAddr), "v"(gOff), "s"(w3p) : "memory");
    }
    if (tid < LQ) qdots[tid] = qdot[b * LQ + tid];
    if (tid < 64) cdots[tid] = cdot[b * LC + ctile * 64 + tid];
    __syncthreads();

    const int     c0   = ctile * 64 + wave * 16;
    const float*  arow = ctx + ((size_t)b * LC + c0 + lm) * DIM; // A row = lane%16
    const __bf16* qb   = Qbf + (size_t)b * LQ * DIM;

    v8f acc[8] = {};   // 8 N-tiles of 16 -> q = 0..127

    for (int k = 0; k < DIM; k += 32) {
        // --- A fragment: (context * w3) -> bf16, contiguous k = 16*half + s ---
        v16bf afr;
        #pragma unroll
        for (int c = 0; c < 2; ++c) {
            const int kb = k + half * 16 + c * 8;
            const float4 x0 = *(const float4*)(arow + kb);
            const float4 x1 = *(const float4*)(arow + kb + 4);
            const float4 g0 = *(const float4*)(&w3s[kb]);
            const float4 g1 = *(const float4*)(&w3s[kb + 4]);
            afr[c * 8 + 0] = tobf(x0.x * g0.x); afr[c * 8 + 1] = tobf(x0.y * g0.y);
            afr[c * 8 + 2] = tobf(x0.z * g0.z); afr[c * 8 + 3] = tobf(x0.w * g0.w);
            afr[c * 8 + 4] = tobf(x1.x * g1.x); afr[c * 8 + 5] = tobf(x1.y * g1.y);
            afr[c * 8 + 6] = tobf(x1.z * g1.z); afr[c * 8 + 7] = tobf(x1.w * g1.w);
        }
        // --- B fragments: pre-converted bf16, contiguous 32B per lane ---
        #pragma unroll
        for (int j = 0; j < 8; ++j) {
            const v16bf bfr =
                *(const v16bf*)(qb + (size_t)(j * 16 + lm) * DIM + k + half * 16);
            acc[j] = __builtin_amdgcn_wmma_f32_16x16x32_bf16(
                false, afr, false, bfr, (short)0, acc[j], false, false);
        }
    }

    // --- bias + softmax over q (row m = r + 8*half lives in acc[*][r]) ---
    #pragma unroll
    for (int r = 0; r < 8; ++r) {
        const float cb = cdots[wave * 16 + r + 8 * half];
        float mx = -3.4e38f;
        #pragma unroll
        for (int j = 0; j < 8; ++j) {
            const float v = acc[j][r] + cb + qdots[j * 16 + lm];
            acc[j][r] = v;
            mx = fmaxf(mx, v);
        }
        for (int off = 1; off < 16; off <<= 1) mx = fmaxf(mx, __shfl_xor(mx, off, 32));
        float s = 0.f;
        #pragma unroll
        for (int j = 0; j < 8; ++j) {
            const float e = __expf(acc[j][r] - mx);
            acc[j][r] = e;
            s += e;
        }
        for (int off = 1; off < 16; off <<= 1) s += __shfl_xor(s, off, 32);
        const float inv = 1.f / s;
        #pragma unroll
        for (int j = 0; j < 8; ++j) acc[j][r] *= inv;
        if (lm == 0) rowmax[b * LC + c0 + r + 8 * half] = mx;  // row max of sim
    }

    // --- store probabilities as bf16, row-major (c, q) ---
    __bf16* Pr = P + ((size_t)b * LC + c0) * LQ;
    #pragma unroll
    for (int r = 0; r < 8; ++r)
        #pragma unroll
        for (int j = 0; j < 8; ++j)
            Pr[(size_t)(r + 8 * half) * LQ + j * 16 + lm] = tobf(acc[j][r]);
}

// ---------------------------------------------------------------------------
// K3: c2q = P @ Q.  A = P (bf16, direct A-layout loads).  B from QbfT[b][d][q]
// so K (=q) is contiguous in memory: every fragment is a plain vector load.
// Block: 128 threads = 4 waves; wave = 16 rows, block.y = 256-wide N quarter.
// ---------------------------------------------------------------------------
__global__ __launch_bounds__(128)
void k_pv(const __bf16* __restrict__ P, const __bf16* __restrict__ QbfT,
          float* __restrict__ c2q)
{
    const int ctile = blockIdx.x;   // 0..15
    const int nq    = blockIdx.y;   // 0..3
    const int b     = blockIdx.z;   // 0..31
    const int tid   = threadIdx.x;
    const int wave  = tid >> 5;
    const int lane  = tid & 31;
    const int half  = lane >> 4;
    const int lm    = lane & 15;

    const int c0 = ctile * 64 + wave * 16;
    const int n0 = nq * 256;

    const __bf16* prow = P + ((size_t)b * LC + c0 + lm) * LQ;
    const __bf16* qT   = QbfT + (size_t)b * DIM * LQ;   // [d][q]

    v8f acc[16] = {};  // 16 N-tiles -> d = n0 .. n0+255

    #pragma unroll
    for (int ks = 0; ks < 4; ++ks) {
        const int k0 = ks * 32;

        const v8bf alo = *(const v8bf*)(prow + k0 + half * 16);
        const v8bf ahi = *(const v8bf*)(prow + k0 + half * 16 + 8);
        v16bf afr;
        #pragma unroll
        for (int e = 0; e < 8; ++e) { afr[e] = alo[e]; afr[8 + e] = ahi[e]; }

        #pragma unroll
        for (int j = 0; j < 16; ++j) {
            // B[k][n] = Q[q=k][d=n] = QbfT[n][k] : contiguous 16 bf16
            const v16bf bfr =
                *(const v16bf*)(qT + (size_t)(n0 + j * 16 + lm) * LQ + k0 + half * 16);
            acc[j] = __builtin_amdgcn_wmma_f32_16x16x32_bf16(
                false, afr, false, bfr, (short)0, acc[j], false, false);
        }
    }

    float* orow = c2q + ((size_t)b * LC + c0) * DIM + n0;
    #pragma unroll
    for (int r = 0; r < 8; ++r)
        #pragma unroll
        for (int j = 0; j < 16; ++j)
            orow[(size_t)(r + 8 * half) * DIM + j * 16 + lm] = acc[j][r];
}

// ---------------------------------------------------------------------------
// K4a: batt[b,c] = softmax_c(rowmax[b,:]).  One block per batch (tiny).
// ---------------------------------------------------------------------------
__global__ __launch_bounds__(256)
void k_batt(const float* __restrict__ rowmax, float* __restrict__ batt)
{
    const int b   = blockIdx.x;
    const int tid = threadIdx.x;
    __shared__ float red[256];

    float v[4];
    float mx = -3.4e38f;
    #pragma unroll
    for (int i = 0; i < 4; ++i) {
        v[i] = rowmax[b * LC + tid + 256 * i];
        mx = fmaxf(mx, v[i]);
    }
    red[tid] = mx; __syncthreads();
    for (int s = 128; s > 0; s >>= 1) {
        if (tid < s) red[tid] = fmaxf(red[tid], red[tid + s]);
        __syncthreads();
    }
    mx = red[0]; __syncthreads();

    float e[4];
    float ls = 0.f;
    #pragma unroll
    for (int i = 0; i < 4; ++i) { e[i] = __expf(v[i] - mx); ls += e[i]; }
    red[tid] = ls; __syncthreads();
    for (int s = 128; s > 0; s >>= 1) {
        if (tid < s) red[tid] += red[tid + s];
        __syncthreads();
    }
    const float inv = 1.f / red[0];
    #pragma unroll
    for (int i = 0; i < 4; ++i) batt[b * LC + tid + 256 * i] = e[i] * inv;
}

// ---------------------------------------------------------------------------
// K4b: partial q2c sums over 256-row c-slabs.
// grid (d-chunk:4, c-slab:4, b:32) = 512 blocks; all loads coalesced.
// ---------------------------------------------------------------------------
__global__ __launch_bounds__(256)
void k_q2c_part(const float* __restrict__ ctx, const float* __restrict__ batt,
                float* __restrict__ part)
{
    const int d     = blockIdx.x * 256 + threadIdx.x;
    const int cbase = blockIdx.y * 256;
    const int b     = blockIdx.z;

    const float* cb = ctx + ((size_t)b * LC + cbase) * DIM + d;
    const float* bw = batt + b * LC + cbase;

    float s = 0.f;
    #pragma unroll 4
    for (int c = 0; c < 256; ++c) s += bw[c] * cb[(size_t)c * DIM];

    part[((size_t)b * 4 + blockIdx.y) * DIM + d] = s;
}

// ---------------------------------------------------------------------------
// K4c: reduce the 4 partials and broadcast q2c over all Lc rows.
// grid (row-group:32, b:32) = 1024 blocks; coalesced stores.
// ---------------------------------------------------------------------------
__global__ __launch_bounds__(256)
void k_q2c_bcast(const float* __restrict__ part, float* __restrict__ out2)
{
    const int b   = blockIdx.y;
    const int r0  = blockIdx.x * 32;     // 32 context rows per block
    const int tid = threadIdx.x;

    float v[4];
    #pragma unroll
    for (int i = 0; i < 4; ++i) {
        const int d = tid + 256 * i;
        float s = 0.f;
        #pragma unroll
        for (int sl = 0; sl < 4; ++sl)
            s += part[((size_t)b * 4 + sl) * DIM + d];
        v[i] = s;
    }

    float* ob = out2 + ((size_t)b * LC + r0) * DIM;
    for (int r = 0; r < 32; ++r)
        #pragma unroll
        for (int i = 0; i < 4; ++i)
            ob[(size_t)r * DIM + tid + 256 * i] = v[i];
}

// ---------------------------------------------------------------------------
extern "C" void kernel_launch(void* const* d_in, const int* in_sizes, int n_in,
                              void* d_out, int out_size, void* d_ws, size_t ws_size,
                              hipStream_t stream)
{
    const float* ctx = (const float*)d_in[0];  // (32,1024,1024) f32
    const float* qry = (const float*)d_in[1];  // (32,128,1024)  f32
    const float* w   = (const float*)d_in[2];  // (3072,)        f32

    float* c2q  = (float*)d_out;                       // output 0
    float* out2 = c2q + (size_t)B_ * LC * DIM;         // output 1 (broadcast q2c)

    // workspace layout (~26 MB):
    float*  qdot = (float*)d_ws;                 // 4096 f32
    float*  cdot = qdot + B_ * LQ;               // 32768 f32
    float*  rmax = cdot + B_ * LC;               // 32768 f32
    float*  batt = rmax + B_ * LC;               // 32768 f32
    float*  part = batt + B_ * LC;               // 32*4*1024 f32 (512 KB)
    __bf16* P    = (__bf16*)(part + B_ * 4 * DIM);   // 32*1024*128 bf16 (8 MB)
    __bf16* Qbf  = P + (size_t)B_ * LC * LQ;         // 32*128*1024 bf16 (8 MB)
    __bf16* QbfT = Qbf + (size_t)B_ * LQ * DIM;      // 32*1024*128 bf16 (8 MB)

    k_rowdots<<<dim3((B_ * LC + B_ * LQ) * 4 / 256), dim3(256), 0, stream>>>(
        ctx, qry, w, cdot, qdot);
    k_convq<<<dim3(DIM / 256, B_), dim3(256), 0, stream>>>(qry, Qbf, QbfT);
    k_sim<<<dim3(LC / 64, B_), dim3(128), 0, stream>>>(
        ctx, Qbf, w, cdot, qdot, rmax, P);
    k_pv<<<dim3(LC / 64, DIM / 256, B_), dim3(128), 0, stream>>>(
        P, QbfT, c2q);
    k_batt<<<dim3(B_), dim3(256), 0, stream>>>(rmax, batt);
    k_q2c_part<<<dim3(DIM / 256, LC / 256, B_), dim3(256), 0, stream>>>(
        ctx, batt, part);
    k_q2c_bcast<<<dim3(LC / 32, B_), dim3(256), 0, stream>>>(part, out2);

    (void)in_sizes; (void)n_in; (void)out_size; (void)ws_size;
}
